// MultiHeadAttention_18949395710401
// MI455X (gfx1250) — matrix operand order repaired and checked by reading, not compiled, not run on hardware
//
#include <hip/hip_runtime.h>

typedef _Float16 v16h __attribute__((ext_vector_type(16)));
typedef _Float16 v8h  __attribute__((ext_vector_type(8)));
typedef float    v8f  __attribute__((ext_vector_type(8)));
typedef float    v4f  __attribute__((ext_vector_type(4)));

#define SEQ   1024
#define EMB   768
#define NH    12
#define ROWS  2048                     // B*S
#define SCALE 0.35355339059327373f    // 1/sqrt(sub=8)
#define LDSW  72                       // padded LDS row stride in halves (144B)

static __device__ __forceinline__ v16h cat8(v8h lo, v8h hi) {
    return __builtin_shufflevector(lo, hi,
           0,1,2,3,4,5,6,7,8,9,10,11,12,13,14,15);
}

static __device__ __forceinline__ v16h cvt16(v4f a, v4f b, v4f c, v4f d) {
    v16h r;
    #pragma unroll
    for (int t = 0; t < 4; ++t) {
        r[t]      = (_Float16)a[t];
        r[t + 4]  = (_Float16)b[t];
        r[t + 8]  = (_Float16)c[t];
        r[t + 12] = (_Float16)d[t];
    }
    return r;
}

// ---------------------------------------------------------------------------
// Kernel 1: P = X @ W^T + bias (f32 in, f16 out); q-path folds lin_w*scale.
// One wave = 32x32 output tile (2x2 WMMA frags) -> 2x operand reuse vs 16x16.
// ---------------------------------------------------------------------------
__global__ __launch_bounds__(256) void mha_proj_f16(
    const float* __restrict__ kin, const float* __restrict__ qin,
    const float* __restrict__ Wq,  const float* __restrict__ bq,
    const float* __restrict__ Wk,  const float* __restrict__ bk,
    const float* __restrict__ lin_w,
    _Float16* __restrict__ qws, _Float16* __restrict__ kws)
{
    const int lane = threadIdx.x & 31;
    const int wave = threadIdx.x >> 5;
    const bool isQ = (blockIdx.z == 0);

    const float* X    = isQ ? qin : kin;
    const float* W    = isQ ? Wq  : Wk;
    const float* bias = isQ ? bq  : bk;
    _Float16*    outp = isQ ? qws : kws;

    const int i0 = blockIdx.x * 32;                  // row strip (0..2016)
    const int o0 = (blockIdx.y * 8 + wave) * 32;     // col strip (0..736)

    const int mrow = lane & 15;
    const int ahi  = (lane >> 4) ? 16 : 0;           // A: upper lanes K+16
    const int bhi  = (lane >> 4) ? 16 : 0;           // B: upper lanes K+16

    const float* xp0 = X + (size_t)(i0 + mrow) * EMB;
    const float* xp1 = xp0 + (size_t)16 * EMB;
    const float* wp0 = W + (size_t)(o0 + mrow) * EMB;
    const float* wp1 = wp0 + (size_t)16 * EMB;

    v8f c00 = {}, c01 = {}, c10 = {}, c11 = {};
    for (int e0 = 0; e0 < EMB; e0 += 32) {
        v16h a0 = cvt16(*(const v4f*)(xp0 + e0 + ahi),
                        *(const v4f*)(xp0 + e0 + ahi + 4),
                        *(const v4f*)(xp0 + e0 + ahi + 8),
                        *(const v4f*)(xp0 + e0 + ahi + 12));
        v16h a1 = cvt16(*(const v4f*)(xp1 + e0 + ahi),
                        *(const v4f*)(xp1 + e0 + ahi + 4),
                        *(const v4f*)(xp1 + e0 + ahi + 8),
                        *(const v4f*)(xp1 + e0 + ahi + 12));
        v16h b0 = cvt16(*(const v4f*)(wp0 + e0 + bhi),
                        *(const v4f*)(wp0 + e0 + bhi + 4),
                        *(const v4f*)(wp0 + e0 + bhi + 8),
                        *(const v4f*)(wp0 + e0 + bhi + 12));
        v16h b1 = cvt16(*(const v4f*)(wp1 + e0 + bhi),
                        *(const v4f*)(wp1 + e0 + bhi + 4),
                        *(const v4f*)(wp1 + e0 + bhi + 8),
                        *(const v4f*)(wp1 + e0 + bhi + 12));

        c00 = __builtin_amdgcn_wmma_f32_16x16x32_f16(false, a0, false, b0,
                                                     (short)0, c00, false, false);
        c01 = __builtin_amdgcn_wmma_f32_16x16x32_f16(false, a0, false, b1,
                                                     (short)0, c01, false, false);
        c10 = __builtin_amdgcn_wmma_f32_16x16x32_f16(false, a1, false, b0,
                                                     (short)0, c10, false, false);
        c11 = __builtin_amdgcn_wmma_f32_16x16x32_f16(false, a1, false, b1,
                                                     (short)0, c11, false, false);
    }

    const int rlo = (lane >> 4) ? 8 : 0;
    #pragma unroll
    for (int ni = 0; ni < 2; ++ni) {
        const int   col = o0 + ni * 16 + mrow;
        const float bv  = bias[col];
        const float mul = isQ ? lin_w[(col & 63) >> 3] * SCALE : 1.0f;
        #pragma unroll
        for (int mi = 0; mi < 2; ++mi) {
            v8f c = (ni == 0) ? (mi == 0 ? c00 : c10)
                              : (mi == 0 ? c01 : c11);
            const size_t rbase = (size_t)(i0 + mi * 16 + rlo);
            #pragma unroll
            for (int r = 0; r < 8; ++r)
                outp[(rbase + r) * EMB + col] = (_Float16)((c[r] + bv) * mul);
        }
    }
}

// ---------------------------------------------------------------------------
// Kernel 2: out[b,h,i,j] = sum_c q'[i][c] * k'[j][c] + lin_b
// Block = 128(M) x 64(N). B-panel (64 rows x 64 ch f16 = 8KB) is staged once
// into LDS with gfx1250 async copies (global_load_async_to_lds_b128,
// ASYNCcnt), then all 8 waves read it via ds_load.
// ---------------------------------------------------------------------------
__global__ __launch_bounds__(256) void mha_score_f16(
    const _Float16* __restrict__ qws, const _Float16* __restrict__ kws,
    const float* __restrict__ lin_b, float* __restrict__ out)
{
    __shared__ _Float16 Bs[64 * LDSW];   // 9216 B, rows padded to 144B

    const int lane = threadIdx.x & 31;
    const int wave = threadIdx.x >> 5;
    const int bh = blockIdx.z;              // 0..23
    const int b  = bh / NH, h = bh % NH;
    const int i0 = blockIdx.x * 128 + wave * 16;
    const int j0 = blockIdx.y * 64;

    const int mrow = lane & 15;
    const int ahi  = (lane >> 4) ? 16 : 0;
    const int bhi  = (lane >> 4) ? 16 : 0;

    // ---- async-stage the shared B panel: k' rows j0..j0+63, 64 channels ----
    const _Float16* kbase = kws + (size_t)(b * SEQ + j0) * EMB + h * 64;
    #pragma unroll
    for (int it = 0; it < 2; ++it) {
        const int c    = threadIdx.x + it * 256;   // 0..511 16B-chunks
        const int row  = c >> 3;                   // 0..63
        const int colh = (c & 7) * 8;              // half offset 0..56
        const _Float16* src = kbase + (size_t)row * EMB + colh;
        const uint32_t  dst = (uint32_t)(uintptr_t)&Bs[row * LDSW + colh];
        asm volatile("global_load_async_to_lds_b128 %0, %1, off"
                     :
                     : "v"(dst), "v"(src)
                     : "memory");
    }
    asm volatile("s_wait_asynccnt 0" ::: "memory");
    __syncthreads();

    // ---- A frag: q' rows i0..i0+15, channels h*64+0..63 (two K=32 chunks) --
    const _Float16* qp = qws + (size_t)(b * SEQ + i0 + mrow) * EMB + h * 64;
    v16h a0 = cat8(*(const v8h*)(qp + ahi),      *(const v8h*)(qp + 8 + ahi));
    v16h a1 = cat8(*(const v8h*)(qp + 32 + ahi), *(const v8h*)(qp + 40 + ahi));

    const float lb = lin_b[0];
    float* obase = out + (size_t)bh * SEQ * SEQ;

    #pragma unroll
    for (int nt = 0; nt < 4; ++nt) {
        const _Float16* bp = &Bs[(nt * 16 + mrow) * LDSW + bhi];
        v16h b0 = *(const v16h*)(bp);        // K = bhi..bhi+15
        v16h b1 = *(const v16h*)(bp + 32);   // K = 32+bhi..+15

        v8f c = {};
        c = __builtin_amdgcn_wmma_f32_16x16x32_f16(false, a0, false, b0,
                                                   (short)0, c, false, false);
        c = __builtin_amdgcn_wmma_f32_16x16x32_f16(false, a1, false, b1,
                                                   (short)0, c, false, false);

        float* op = obase + (size_t)(i0 + ((lane >> 4) ? 8 : 0)) * SEQ
                          + j0 + nt * 16 + mrow;
        #pragma unroll
        for (int r = 0; r < 8; ++r)
            op[(size_t)r * SEQ] = c[r] + lb;
    }
}

// ---------------------------------------------------------------------------
extern "C" void kernel_launch(void* const* d_in, const int* in_sizes, int n_in,
                              void* d_out, int out_size, void* d_ws, size_t ws_size,
                              hipStream_t stream) {
    // setup_inputs order: k, q, Wq_w, Wq_b, Wk_w, Wk_b, lin_w, lin_b
    const float* kin   = (const float*)d_in[0];
    const float* qin   = (const float*)d_in[1];
    const float* Wq    = (const float*)d_in[2];
    const float* bq    = (const float*)d_in[3];
    const float* Wk    = (const float*)d_in[4];
    const float* bk    = (const float*)d_in[5];
    const float* lin_w = (const float*)d_in[6];
    const float* lin_b = (const float*)d_in[7];
    float* out = (float*)d_out;

    _Float16* qws = (_Float16*)d_ws;                   // [2048][768] f16
    _Float16* kws = qws + (size_t)ROWS * EMB;          // [2048][768] f16

    // Stage 1: projections, 32x32 tile per wave.
    // grid: (2048/32) x (24 col-strips / 8 waves) x {q,k}
    dim3 g1(ROWS / 32, (EMB / 32) / 8, 2);
    mha_proj_f16<<<g1, 256, 0, stream>>>(kin, qin, Wq, bq, Wk, bk, lin_w,
                                         qws, kws);

    // Stage 2: batched score GEMM + lin_b; B panel async-staged in LDS.
    // grid: (1024/128) x (1024/64) x 24
    dim3 g2(SEQ / 128, SEQ / 64, 2 * NH);
    mha_score_f16<<<g2, 256, 0, stream>>>(qws, kws, lin_b, out);
}
